// LongformerSelfAttention_12730283065484
// MI455X (gfx1250) — matrix operand order repaired and checked by reading, not compiled, not run on hardware
//
#include <hip/hip_runtime.h>
#include <hip/hip_bf16.h>

typedef _Float16 half_t;
typedef __attribute__((ext_vector_type(16))) _Float16 v16h;
typedef __attribute__((ext_vector_type(8)))  _Float16 v8h;
typedef __attribute__((ext_vector_type(8)))  float    v8f;

union AFrag { v16h v; v8h h[2]; };

static __device__ __forceinline__ v8f wmma_f16(v16h a, v16h b, v8f c) {
    // D = A(16x32 f16) * B(32x16 f16) + C(16x16 f32)
    return __builtin_amdgcn_wmma_f32_16x16x32_f16(
        /*neg_a=*/false, a, /*neg_b=*/false, b,
        /*c_mod=*/(short)0, c, /*reuse_a=*/false, /*reuse_b=*/false);
}

#define SEQ   4096
#define EMB   768
#define NHEAD 12
#define HDIM  64
#define WSIDE 256

// ---------------------------------------------------------------------------
// Phase 0: fp32 -> f16 conversion
// ---------------------------------------------------------------------------
__global__ void cvt_f32_to_f16(const float* __restrict__ src,
                               half_t* __restrict__ dst, int n) {
    int i = blockIdx.x * blockDim.x + threadIdx.x;
    if (i < n) dst[i] = (half_t)src[i];
}

// ---------------------------------------------------------------------------
// Phase 1: fused QKV projection GEMM.
//   out[m,n] = dot(hs[m,:], W[n,:]) + bias[n]   (W row-major, K contiguous)
//   z = blockIdx.z selects Q (scaled by 1/8) / K / V.
//   Q,K stored as [BH][S][64] f16; V stored transposed [BH][64][S] f16.
//   One wave computes a 16x64 strip (4 accumulators, shared A fragment).
// ---------------------------------------------------------------------------
__global__ __launch_bounds__(128) void qkv_gemm_kernel(
    const half_t* __restrict__ A,
    const half_t* __restrict__ Wq, const half_t* __restrict__ Wk,
    const half_t* __restrict__ Wv,
    const float* __restrict__ bq, const float* __restrict__ bk,
    const float* __restrict__ bv,
    half_t* __restrict__ Qo, half_t* __restrict__ Ko, half_t* __restrict__ Vt)
{
    const int z    = blockIdx.z;
    const half_t* W   = (z == 0) ? Wq : (z == 1) ? Wk : Wv;
    const float*  bia = (z == 0) ? bq : (z == 1) ? bk : bv;

    const int lane = threadIdx.x & 31;
    const int wave = threadIdx.x >> 5;
    const int lo   = lane & 15;
    const int hi   = lane >> 4;
    const int mt   = blockIdx.x * 4 + wave;   // 0..511 row tiles (16 rows each)
    const int m0   = mt * 16;

    v8f acc[4];
    #pragma unroll
    for (int j = 0; j < 4; ++j) acc[j] = (v8f){};

    const half_t* arow = A + (size_t)(m0 + lo) * EMB;

    for (int kk = 0; kk < EMB; kk += 32) {
        AFrag a;
        a.h[0] = *(const v8h*)(arow + kk + 16 * hi);       // K = 16hi..16hi+7
        a.h[1] = *(const v8h*)(arow + kk + 8 + 16 * hi);   // K = 16hi+8..16hi+15
        #pragma unroll
        for (int j = 0; j < 4; ++j) {
            int n = (blockIdx.y * 4 + j) * 16 + lo;        // B column = W row n
            v16h bf = *(const v16h*)(W + (size_t)n * EMB + kk + 16 * hi);
            acc[j] = wmma_f16(a.v, bf, acc[j]);
        }
    }

    #pragma unroll
    for (int j = 0; j < 4; ++j) {
        int n  = (blockIdx.y * 4 + j) * 16 + lo;
        float bval = bia[n];
        int h = n >> 6, d = n & 63;
        if (z < 2) {
            half_t* dst = (z == 0) ? Qo : Ko;
            #pragma unroll
            for (int r = 0; r < 8; ++r) {
                int m = m0 + r + 8 * hi;
                int b = m >> 12, s = m & (SEQ - 1);
                float v = acc[j][r] + bval;
                if (z == 0) v *= 0.125f;   // 1/sqrt(64)
                dst[(((size_t)(b * NHEAD + h)) * SEQ + s) * HDIM + d] = (half_t)v;
            }
        } else {
            // V transposed: [BH][64][S]; lane's 8 rows are 8 consecutive s.
            int b  = m0 >> 12;
            int sb = (m0 + 8 * hi) & (SEQ - 1);
            v8h pack;
            #pragma unroll
            for (int r = 0; r < 8; ++r) pack[r] = (half_t)(acc[j][r] + bval);
            *(v8h*)(Vt + (((size_t)(b * NHEAD + h)) * HDIM + d) * SEQ + sb) = pack;
        }
    }
}

// ---------------------------------------------------------------------------
// Phase 2: sliding-window flash attention.
//   One wave per (bh, 16-query block). 17 key tiles of 32 cover the band.
// ---------------------------------------------------------------------------
__global__ __launch_bounds__(128) void sw_attn_kernel(
    const half_t* __restrict__ Q, const half_t* __restrict__ K,
    const half_t* __restrict__ Vt, float* __restrict__ out)
{
    __shared__ __align__(32) half_t plds_all[4 * 16 * 32];

    const int lane = threadIdx.x & 31;
    const int wave = threadIdx.x >> 5;
    const int lo   = lane & 15;
    const int hi   = lane >> 4;
    const int qs   = (blockIdx.x * 4 + wave) * 16;  // query block start
    const int bh   = blockIdx.y;
    const int b    = bh / NHEAD, h = bh % NHEAD;

    const half_t* Qp = Q  + (size_t)bh * SEQ * HDIM;
    const half_t* Kp = K  + (size_t)bh * SEQ * HDIM;
    const half_t* Vp = Vt + (size_t)bh * HDIM * SEQ;
    half_t* plds = plds_all + wave * 512;

    // Q A-fragments (d split into two K=32 chunks); q already pre-scaled.
    AFrag aq0, aq1;
    const half_t* qrow = Qp + (size_t)(qs + lo) * HDIM;
    aq0.h[0] = *(const v8h*)(qrow + 16 * hi);
    aq0.h[1] = *(const v8h*)(qrow + 8 + 16 * hi);
    aq1.h[0] = *(const v8h*)(qrow + 32 + 16 * hi);
    aq1.h[1] = *(const v8h*)(qrow + 40 + 16 * hi);

    v8f O[4];
    #pragma unroll
    for (int dt = 0; dt < 4; ++dt) O[dt] = (v8f){};
    float mrow[8], lrow[8];
    #pragma unroll
    for (int r = 0; r < 8; ++r) { mrow[r] = -1e30f; lrow[r] = 0.f; }
    const v8f zero8 = (v8f){};

    for (int t = 0; t < 17; ++t) {
        const int kb = qs - WSIDE + t * 32;
        const int k0 = kb + lo, k1 = k0 + 16;
        const int k0c = min(max(k0, 0), SEQ - 1);
        const int k1c = min(max(k1, 0), SEQ - 1);

        // K B-fragments: column n = key row; contiguous 32B chunks per d half.
        const half_t* kr0 = Kp + (size_t)k0c * HDIM + 16 * hi;
        const half_t* kr1 = Kp + (size_t)k1c * HDIM + 16 * hi;
        v16h b00 = *(const v16h*)(kr0);        // d 0..31
        v16h b01 = *(const v16h*)(kr0 + 32);   // d 32..63
        v16h b10 = *(const v16h*)(kr1);
        v16h b11 = *(const v16h*)(kr1 + 32);

        v8f s0 = wmma_f16(aq0.v, b00, wmma_f16(aq1.v, b01, zero8));
        v8f s1 = wmma_f16(aq0.v, b10, wmma_f16(aq1.v, b11, zero8));

        // Band + bounds mask, per-row running max.
        float tmax[8];
        #pragma unroll
        for (int r = 0; r < 8; ++r) {
            int qi = qs + r + 8 * hi;
            bool v0 = (k0 >= 0) && (k0 < SEQ) && (k0 >= qi - WSIDE) && (k0 <= qi + WSIDE);
            bool v1 = (k1 >= 0) && (k1 < SEQ) && (k1 >= qi - WSIDE) && (k1 <= qi + WSIDE);
            s0[r] = v0 ? s0[r] : -1e30f;
            s1[r] = v1 ? s1[r] : -1e30f;
            tmax[r] = fmaxf(s0[r], s1[r]);
        }
        #pragma unroll
        for (int r = 0; r < 8; ++r)
            #pragma unroll
            for (int msk = 1; msk < 16; msk <<= 1)
                tmax[r] = fmaxf(tmax[r], __shfl_xor(tmax[r], msk, 32));

        float scal[8];
        #pragma unroll
        for (int r = 0; r < 8; ++r) {
            float mn = fmaxf(mrow[r], tmax[r]);
            scal[r] = __expf(mrow[r] - mn);
            mrow[r] = mn;
        }

        // Probabilities -> LDS (C-layout -> A-layout relayout), row sums.
        float psum[8];
        #pragma unroll
        for (int r = 0; r < 8; ++r) {
            float p0 = __expf(s0[r] - mrow[r]);
            float p1 = __expf(s1[r] - mrow[r]);
            int m = r + 8 * hi;
            plds[m * 32 + lo]      = (half_t)p0;
            plds[m * 32 + 16 + lo] = (half_t)p1;
            psum[r] = p0 + p1;
        }
        #pragma unroll
        for (int r = 0; r < 8; ++r)
            #pragma unroll
            for (int msk = 1; msk < 16; msk <<= 1)
                psum[r] += __shfl_xor(psum[r], msk, 32);

        #pragma unroll
        for (int r = 0; r < 8; ++r)
            lrow[r] = lrow[r] * scal[r] + psum[r];
        #pragma unroll
        for (int dt = 0; dt < 4; ++dt)
            #pragma unroll
            for (int r = 0; r < 8; ++r)
                O[dt][r] *= scal[r];

        // P A-fragment from LDS (same-wave LDS ops are in order).
        AFrag ap;
        ap.h[0] = *(const v8h*)(plds + lo * 32 + 16 * hi);
        ap.h[1] = *(const v8h*)(plds + lo * 32 + 8 + 16 * hi);

        // V B-fragments: Vt rows are contiguous over s (= K dim of PV).
        int sb = min(max(kb + 16 * hi, 0), SEQ - 16);
        #pragma unroll
        for (int dt = 0; dt < 4; ++dt) {
            v16h bv = *(const v16h*)(Vp + (size_t)(dt * 16 + lo) * SEQ + sb);
            O[dt] = wmma_f16(ap.v, bv, O[dt]);
        }
    }

    // Normalize and write [B,S,E] fp32, E order = h*64 + d.
    float inv[8];
    #pragma unroll
    for (int r = 0; r < 8; ++r) inv[r] = 1.0f / lrow[r];
    #pragma unroll
    for (int dt = 0; dt < 4; ++dt)
        #pragma unroll
        for (int r = 0; r < 8; ++r) {
            int s = qs + r + 8 * hi;
            out[((size_t)(b * SEQ + s)) * EMB + h * HDIM + dt * 16 + lo] =
                O[dt][r] * inv[r];
        }
}

// ---------------------------------------------------------------------------
extern "C" void kernel_launch(void* const* d_in, const int* in_sizes, int n_in,
                              void* d_out, int out_size, void* d_ws, size_t ws_size,
                              hipStream_t stream) {
    (void)in_sizes; (void)n_in; (void)out_size; (void)ws_size;
    const float* hs = (const float*)d_in[0];
    const float* qw = (const float*)d_in[1];
    const float* qb = (const float*)d_in[2];
    const float* kw = (const float*)d_in[3];
    const float* kb = (const float*)d_in[4];
    const float* vw = (const float*)d_in[5];
    const float* vb = (const float*)d_in[6];
    float* out = (float*)d_out;

    const int NHS = 2 * SEQ * EMB;   // 6,291,456
    const int NW  = EMB * EMB;       //   589,824
    half_t* ws   = (half_t*)d_ws;
    half_t* hs16 = ws;
    half_t* qw16 = hs16 + NHS;
    half_t* kw16 = qw16 + NW;
    half_t* vw16 = kw16 + NW;
    half_t* Q16  = vw16 + NW;        // [24][4096][64]
    half_t* K16  = Q16 + NHS;
    half_t* Vt16 = K16 + NHS;        // [24][64][4096]

    cvt_f32_to_f16<<<(NHS + 255) / 256, 256, 0, stream>>>(hs, hs16, NHS);
    cvt_f32_to_f16<<<(NW + 255) / 256, 256, 0, stream>>>(qw, qw16, NW);
    cvt_f32_to_f16<<<(NW + 255) / 256, 256, 0, stream>>>(kw, kw16, NW);
    cvt_f32_to_f16<<<(NW + 255) / 256, 256, 0, stream>>>(vw, vw16, NW);

    // 512 row tiles / 4 waves per block; 12 col-strips; z = Q/K/V
    qkv_gemm_kernel<<<dim3(128, 12, 3), 128, 0, stream>>>(
        hs16, qw16, kw16, vw16, qb, kb, vb, Q16, K16, Vt16);

    // 256 query blocks / 4 waves per block; 24 (b,h) pairs
    sw_attn_kernel<<<dim3(64, 24), 128, 0, stream>>>(Q16, K16, Vt16, out);
}
